// Int8Linear_549755814067
// MI455X (gfx1250) — 8-bit matrix operand order repaired and checked by reading, compiled, not run on hardware
//
#include <hip/hip_runtime.h>
#include <stdint.h>

typedef __attribute__((ext_vector_type(8))) int v8i;
typedef __attribute__((ext_vector_type(4))) int v4i_t;

#define B_   8
#define S_   2048
#define IN_  1024
#define OUT_ 4096
#define M_   (B_ * S_)
#define K_   IN_
#define N_   OUT_

#define TM 128
#define TN 128
#define TK 64
#define KSTEPS (K_ / TK)

#if defined(__has_builtin)
#if __has_builtin(__builtin_amdgcn_global_load_async_to_lds_b128) && \
    __has_builtin(__builtin_amdgcn_s_wait_asynccnt)
#define USE_ASYNC 1
#endif
#endif

#if defined(USE_ASYNC)
typedef __attribute__((address_space(1))) v4i_t g_v4i;
typedef __attribute__((address_space(3))) v4i_t l_v4i;
__device__ __forceinline__ void async_cp16(const signed char* g, signed char* l) {
  __builtin_amdgcn_global_load_async_to_lds_b128((g_v4i*)g, (l_v4i*)l, 0, 0);
}
__device__ __forceinline__ void async_wait0() { __builtin_amdgcn_s_wait_asynccnt(0); }
#else
__device__ __forceinline__ void async_cp16(const signed char* g, signed char* l) {
  *(int4*)l = *(const int4*)g;
}
__device__ __forceinline__ void async_wait0() {}
#endif

__device__ __forceinline__ unsigned encf(float f) {
  unsigned u = __float_as_uint(f);
  return (u & 0x80000000u) ? ~u : (u | 0x80000000u);
}
__device__ __forceinline__ float decf(unsigned e) {
  unsigned u = (e & 0x80000000u) ? (e & 0x7fffffffu) : ~e;
  return __uint_as_float(u);
}

__global__ void init_k(unsigned* enc) {
  enc[0] = 0u;
  enc[1] = 0xffffffffu;
}

__global__ __launch_bounds__(256) void minmax_k(const float* __restrict__ x,
                                                unsigned* __restrict__ enc) {
  __shared__ float smx[256], smn[256];
  int t = threadIdx.x;
  size_t base = ((size_t)blockIdx.x * 256 + t) * 32;
  float mx = -3.4e38f, mn = 3.4e38f;
#pragma unroll
  for (int i = 0; i < 8; i++) {
    float4 v = *(const float4*)(x + base + (size_t)i * 4);
    mx = fmaxf(mx, fmaxf(fmaxf(v.x, v.y), fmaxf(v.z, v.w)));
    mn = fminf(mn, fminf(fminf(v.x, v.y), fminf(v.z, v.w)));
  }
  smx[t] = mx; smn[t] = mn;
  __syncthreads();
  for (int s = 128; s > 0; s >>= 1) {
    if (t < s) { smx[t] = fmaxf(smx[t], smx[t + s]); smn[t] = fminf(smn[t], smn[t + s]); }
    __syncthreads();
  }
  if (t == 0) {
    atomicMax(&enc[0], encf(smx[0]));
    atomicMin(&enc[1], encf(smn[0]));
  }
}

__global__ void scale_k(const unsigned* __restrict__ enc, float* __restrict__ sc) {
  float mx = decf(enc[0]), mn = decf(enc[1]);
  float scale = (mx - mn) / 255.0f;
  float zp = rintf((mx * -128.0f - mn * 127.0f) / (mx - mn));
  sc[0] = scale;
  sc[1] = zp;
}

__global__ __launch_bounds__(256) void quant_k(const float* __restrict__ x,
                                               const float* __restrict__ sc,
                                               signed char* __restrict__ x8,
                                               int* __restrict__ xsum) {
  __shared__ int ssum[256];
  int row = blockIdx.x, t = threadIdx.x;
  float scale = sc[0], zp = sc[1];
  float inv = 1.0f / scale;
  float4 v = *(const float4*)(x + (size_t)row * K_ + t * 4);
  int q0 = (int)(rintf(v.x * inv) + zp);
  int q1 = (int)(rintf(v.y * inv) + zp);
  int q2 = (int)(rintf(v.z * inv) + zp);
  int q3 = (int)(rintf(v.w * inv) + zp);
  int packed = (q0 & 0xff) | ((q1 & 0xff) << 8) | ((q2 & 0xff) << 16) | ((q3 & 0xff) << 24);
  *(int*)(x8 + (size_t)row * K_ + t * 4) = packed;
  int lsum = (int)(signed char)q0 + (int)(signed char)q1 +
             (int)(signed char)q2 + (int)(signed char)q3;
  ssum[t] = lsum;
  __syncthreads();
  for (int s = 128; s > 0; s >>= 1) {
    if (t < s) ssum[t] += ssum[t + s];
    __syncthreads();
  }
  if (t == 0) xsum[row] = ssum[0];
}

__global__ __launch_bounds__(256) void packw_k(const int* __restrict__ wq,
                                               signed char* __restrict__ w8,
                                               int* __restrict__ wsum) {
  __shared__ int ssum[256];
  int row = blockIdx.x, t = threadIdx.x;
  int4 v = *(const int4*)(wq + (size_t)row * K_ + t * 4);
  int packed = (v.x & 0xff) | ((v.y & 0xff) << 8) | ((v.z & 0xff) << 16) | ((v.w & 0xff) << 24);
  *(int*)(w8 + (size_t)row * K_ + t * 4) = packed;
  ssum[t] = v.x + v.y + v.z + v.w;
  __syncthreads();
  for (int s = 128; s > 0; s >>= 1) {
    if (t < s) ssum[t] += ssum[t + s];
    __syncthreads();
  }
  if (t == 0) wsum[row] = ssum[0];
}

__global__ __launch_bounds__(256) void gemm_k(
    const signed char* __restrict__ x8, const signed char* __restrict__ w8,
    const int* __restrict__ xsum, const int* __restrict__ wsum,
    const float* __restrict__ sc, const float* __restrict__ wscale_p,
    const float* __restrict__ wzp_p, const float* __restrict__ bias,
    float* __restrict__ out) {
  __shared__ __align__(16) signed char Als[2 * TM * TK];
  __shared__ __align__(16) signed char Bls[2 * TN * TK];

  int t = threadIdx.x;
  int lane = t & 31;
  int wave = t >> 5;
  int wm = wave & 3;
  int wn = wave >> 2;
  int tileM = blockIdx.y * TM;
  int tileN = blockIdx.x * TN;

  int khi = lane >> 4;
  int lm = lane & 15;

  int sr = t >> 1, soff = (t & 1) * 32;
  const signed char* asrc = x8 + (size_t)(tileM + sr) * K_ + soff;
  const signed char* bsrc = w8 + (size_t)(tileN + sr) * K_ + soff;
  signed char* adst = Als + sr * TK + soff;
  signed char* bdst = Bls + sr * TK + soff;

  v8i acc[2][4] = {};

  async_cp16(asrc, adst);
  async_cp16(asrc + 16, adst + 16);
  async_cp16(bsrc, bdst);
  async_cp16(bsrc + 16, bdst + 16);
  async_wait0();
  __syncthreads();

  for (int step = 0; step < KSTEPS; step++) {
    int buf = step & 1;
    if (step + 1 < KSTEPS) {
      int nb = (step + 1) & 1;
      const signed char* ga = asrc + (step + 1) * TK;
      const signed char* gb = bsrc + (step + 1) * TK;
      async_cp16(ga, adst + nb * (TM * TK));
      async_cp16(ga + 16, adst + nb * (TM * TK) + 16);
      async_cp16(gb, bdst + nb * (TN * TK));
      async_cp16(gb + 16, bdst + nb * (TN * TK) + 16);
    }

    const signed char* Ab = Als + buf * (TM * TK);
    const signed char* Bb = Bls + buf * (TN * TK);

    v8i a[2], b[4];
#pragma unroll
    for (int si = 0; si < 2; si++) {
      const signed char* ap = Ab + (wm * 32 + si * 16 + lm) * TK + khi * 16;
#pragma unroll
      for (int p = 0; p < 4; p++) {
        int2 d = *(const int2*)(ap + (p & 1) * 8 + (p >> 1) * 32);
        a[si][2 * p]     = d.x;
        a[si][2 * p + 1] = d.y;
      }
    }
#pragma unroll
    for (int sj = 0; sj < 4; sj++) {
      const signed char* bp = Bb + (wn * 64 + sj * 16 + lm) * TK + khi * 16;
      int4 lo = *(const int4*)(bp);
      int4 hi = *(const int4*)(bp + 32);
      b[sj][0] = lo.x; b[sj][1] = lo.y; b[sj][2] = lo.z; b[sj][3] = lo.w;
      b[sj][4] = hi.x; b[sj][5] = hi.y; b[sj][6] = hi.z; b[sj][7] = hi.w;
    }
#pragma unroll
    for (int si = 0; si < 2; si++)
#pragma unroll
      for (int sj = 0; sj < 4; sj++)
        acc[si][sj] = __builtin_amdgcn_wmma_i32_16x16x64_iu8(
             true, a[si],  true, b[sj], acc[si][sj],
             false,  false);

    if (step + 1 < KSTEPS) {
      async_wait0();
      __syncthreads();
    }
  }

  float xscale = sc[0], xzp = sc[1];
  float s = xscale * wscale_p[0];
  float wzp = wzp_p[0];
  float czz = (float)K_ * xzp * wzp;
#pragma unroll
  for (int si = 0; si < 2; si++) {
    int mBase = tileM + wm * 32 + si * 16 + khi * 8;
#pragma unroll
    for (int sj = 0; sj < 4; sj++) {
      int nIdx = tileN + wn * 64 + sj * 16 + lm;
      float wsn = (float)wsum[nIdx];
      float bnv = bias[nIdx];
#pragma unroll
      for (int r = 0; r < 8; r++) {
        int m = mBase + r;
        float xs = (float)xsum[m];
        float v = ((float)acc[si][sj][r] - wzp * xs - xzp * wsn + czz) * s + bnv;
        __builtin_nontemporal_store(v, out + (size_t)m * N_ + nIdx);
      }
    }
  }
}

extern "C" void kernel_launch(void* const* d_in, const int* in_sizes, int n_in,
                              void* d_out, int out_size, void* d_ws, size_t ws_size,
                              hipStream_t stream) {
  const float* x      = (const float*)d_in[0];
  const int*   wq     = (const int*)d_in[1];
  const float* wscale = (const float*)d_in[2];
  const float* wzp    = (const float*)d_in[3];
  const float* bias   = (const float*)d_in[4];
  float* out = (float*)d_out;

  char* W = (char*)d_ws;
  unsigned* enc = (unsigned*)W;
  float* sc = (float*)(W + 8);
  int* xsum = (int*)(W + 256);
  int* wsum = (int*)(W + 256 + sizeof(int) * M_);
  signed char* x8 = (signed char*)(W + 256 + sizeof(int) * (M_ + N_));
  signed char* w8 = x8 + (size_t)M_ * K_;

  init_k<<<1, 1, 0, stream>>>(enc);
  minmax_k<<<(M_ * K_) / (256 * 32), 256, 0, stream>>>(x, enc);
  scale_k<<<1, 1, 0, stream>>>(enc, sc);
  quant_k<<<M_, 256, 0, stream>>>(x, sc, x8, xsum);
  packw_k<<<N_, 256, 0, stream>>>(wq, w8, wsum);
  gemm_k<<<dim3(N_ / TN, M_ / TM), 256, 0, stream>>>(x8, w8, xsum, wsum, sc,
                                                     wscale, wzp, bias, out);
}
